// S2VTAttModel_58162447123157
// MI455X (gfx1250) — compile-verified, matrix operand order checked by reading, not run on hardware
//
#include <hip/hip_runtime.h>
#include <math.h>

typedef _Float16 h16;
typedef _Float16 v16h __attribute__((ext_vector_type(16)));
typedef _Float16 v8h  __attribute__((ext_vector_type(8)));
typedef float    v8f  __attribute__((ext_vector_type(8)));

#define BB      128
#define TTT     80
#define DVV     512
#define HHH     512
#define VVV     16000
#define MAXLENN 28
#define NSTEP   27
#define NCHUNK  10
#define CLL     8

__device__ __forceinline__ float sigf(float x) { return 1.f / (1.f + __expf(-x)); }

// Load a 16-bit WMMA fragment: this lane's K-pieces [0..7] and [16..23]
// relative to p (caller applies the +8*hsel lane offset). Works for both
// global and LDS pointers.
template <typename P>
__device__ __forceinline__ v16h ldfrag(const P* p) {
  v8h x0 = *(const v8h*)(p);
  v8h x1 = *(const v8h*)(p + 16);
  v16h r;
#pragma unroll
  for (int i = 0; i < 8; ++i) { r[i] = x0[i]; r[i + 8] = x1[i]; }
  return r;
}

__device__ __forceinline__ v8f wmma_f16(v16h a, v16h b, v8f c) {
  return __builtin_amdgcn_wmma_f32_16x16x32_f16(
      /*neg_a=*/false, a, /*neg_b=*/false, b,
      /*c_mod=*/(short)0, c, /*reuse_a=*/false, /*reuse_b=*/false);
}

// ---------------------------------------------------------------------------
// WMMA GEMM: out[m,n] = sum_k A[m,k] * W[n,k] + bias[n]
// A: f16 [M,K] row-major, W: f16 [N,K] row-major (PyTorch weight layout).
// Block = 256 threads = 8 waves arranged 2(M) x 4(N); block tile 64x128.
// Each wave register-blocks a 32x32 tile (2 A-frags x 2 B-frags, 4 wmma per
// 32-wide K-step -> 32 FLOP/B). The 64x32 A tile is staged through LDS once
// per K-chunk and shared by the 4 waves in each M-row (ds_load-fed WMMA).
// Requires M%64==0, N%128==0, K%32==0.
// ---------------------------------------------------------------------------
__global__ __launch_bounds__(256)
void k_gemm_wmma(const h16* __restrict__ A, const h16* __restrict__ W,
                 const float* __restrict__ bias, float* __restrict__ out,
                 int M, int N, int K) {
  __shared__ h16 As[64 * 32];

  const int lane = threadIdx.x & 31;
  const int wave = threadIdx.x >> 5;
  const int l16  = lane & 15;
  const int hsel = lane >> 4;               // which K-half this lane holds
  const int mSub = wave >> 2;               // 0..1
  const int nSub = wave & 3;                // 0..3
  const int mBase = blockIdx.y * 64 + mSub * 32;
  const int nBase = blockIdx.x * 128 + nSub * 32;

  // cooperative A-tile staging coords: 256 thr x 8 halves = 64x32 tile
  const int sR = threadIdx.x >> 2;          // 0..63
  const int sC = (threadIdx.x & 3) * 8;     // 0,8,16,24
  const h16* aStage = A + (size_t)(blockIdx.y * 64 + sR) * K + sC;

  const h16* w0row = W + (size_t)(nBase + l16) * K + 8 * hsel;
  const h16* w1row = W + (size_t)(nBase + 16 + l16) * K + 8 * hsel;

  v8f acc00 = {}, acc01 = {}, acc10 = {}, acc11 = {};

  for (int k0 = 0; k0 < K; k0 += 32) {
    if (k0 + 64 < K) {  // CDNA5 global_prefetch_b8 hints
      __builtin_prefetch(w0row + k0 + 64, 0, 1);
      __builtin_prefetch(w1row + k0 + 64, 0, 1);
      __builtin_prefetch(aStage + k0 + 64, 0, 1);
    }
    // stage A 64x32 tile into LDS (one b128 per thread)
    *(v8h*)&As[sR * 32 + sC] = *(const v8h*)(aStage + k0);
    __syncthreads();

    v16h a0 = ldfrag(&As[(mSub * 32 + l16) * 32 + 8 * hsel]);
    v16h a1 = ldfrag(&As[(mSub * 32 + 16 + l16) * 32 + 8 * hsel]);
    v16h b0 = ldfrag(w0row + k0);
    v16h b1 = ldfrag(w1row + k0);

    acc00 = wmma_f16(a0, b0, acc00);
    acc01 = wmma_f16(a0, b1, acc01);
    acc10 = wmma_f16(a1, b0, acc10);
    acc11 = wmma_f16(a1, b1, acc11);
    __syncthreads();
  }

  // epilogue: C/D layout is VGPR r, lanes 0-15 -> M=r, lanes 16-31 -> M=r+8
  auto store_tile = [&](v8f acc, int mB, int nB) {
    const int n = nB + l16;
    const float bv = bias ? bias[n] : 0.f;
#pragma unroll
    for (int r = 0; r < 8; ++r)
      out[(size_t)(mB + r + 8 * hsel) * N + n] = acc[r] + bv;
  };
  store_tile(acc00, mBase,      nBase);
  store_tile(acc01, mBase,      nBase + 16);
  store_tile(acc10, mBase + 16, nBase);
  store_tile(acc11, mBase + 16, nBase + 16);
}

// --- elementwise helpers ----------------------------------------------------
__global__ void k_pack_f16(h16* __restrict__ dst, const float* __restrict__ src,
                           int rows, int cols, int dstld, int dstoff) {
  int idx = blockIdx.x * blockDim.x + threadIdx.x;
  if (idx >= rows * cols) return;
  int r = idx / cols, c = idx - r * cols;
  dst[(size_t)r * dstld + dstoff + c] = (h16)src[idx];
}

__global__ void k_zero_f32(float* p, int n) {
  int i = blockIdx.x * blockDim.x + threadIdx.x;
  if (i < n) p[i] = 0.f;
}
__global__ void k_zero_f16(h16* p, int n) {
  int i = blockIdx.x * blockDim.x + threadIdx.x;
  if (i < n) p[i] = (h16)0.f;
}

// dst[m, 0:Ka] = a[m*aStride + .], dst[m, Ka:Ka+Kb] = b[m*bStride + .]
__global__ void k_concat2(h16* __restrict__ dst,
                          const h16* __restrict__ a, int aStride,
                          const h16* __restrict__ b, int bStride,
                          int M, int Ka, int Kb) {
  int Kt = Ka + Kb;
  int idx = blockIdx.x * blockDim.x + threadIdx.x;
  if (idx >= M * Kt) return;
  int m = idx / Kt, c = idx - m * Kt;
  dst[idx] = (c < Ka) ? a[(size_t)m * aStride + c]
                      : b[(size_t)m * bStride + (c - Ka)];
}

// LSTM cell pointwise: z[m, 4H] gate order i,f,g,o
__global__ void k_lstm_cell(const float* __restrict__ z, float* __restrict__ c,
                            float* __restrict__ h, h16* __restrict__ h_h, int M) {
  int idx = blockIdx.x * blockDim.x + threadIdx.x;
  if (idx >= M * HHH) return;
  int m = idx / HHH, ch = idx - m * HHH;
  const float* zr = z + (size_t)m * 4 * HHH;
  float zi = zr[ch], zf = zr[HHH + ch], zg = zr[2 * HHH + ch], zo = zr[3 * HHH + ch];
  float cc = sigf(zf) * c[idx] + sigf(zi) * tanhf(zg);
  float hh = sigf(zo) * tanhf(cc);
  c[idx] = cc; h[idx] = hh;
  if (h_h) h_h[idx] = (h16)hh;
}

__global__ void k_snap_ofirst(float* __restrict__ o_first,
                              const float* __restrict__ h, int M) {
  int m = blockIdx.x * blockDim.x + threadIdx.x;
  if (m < M) o_first[m] = h[(size_t)m * HHH];
}

// fil0: first output of the zero-input encoder LSTM (h0=c0=0 => z = b)
__global__ void k_fil0(const float* __restrict__ b, float* __restrict__ fil0) {
  if (threadIdx.x == 0 && blockIdx.x == 0) {
    float zi = b[0], zf = b[HHH], zg = b[2 * HHH], zo = b[3 * HHH];
    float cc = sigf(zf) * 0.f + sigf(zi) * tanhf(zg);
    fil0[0] = sigf(zo) * tanhf(cc);
  }
}

__global__ void k_select_j(const float* __restrict__ o_first,
                           const float* __restrict__ fil0, int* __restrict__ j) {
  int b = blockIdx.x * blockDim.x + threadIdx.x;
  if (b >= BB) return;
  float f0 = fil0[0];
  int cnt = 0;
  for (int k = 0; k < NCHUNK; ++k) cnt += (o_first[b * NCHUNK + k] != f0) ? 1 : 0;
  j[b] = cnt - 1;
}

// h_dec[b,:] = enc_out2[b, j[b], :]  (one-hot einsum; j<0 -> zeros)
__global__ void k_gather_hdec(const float* __restrict__ eo2,
                              const int* __restrict__ j, float* __restrict__ h) {
  int idx = blockIdx.x * blockDim.x + threadIdx.x;
  if (idx >= BB * HHH) return;
  int b = idx / HHH, c = idx - b * HHH;
  int jj = j[b];
  h[idx] = (jj >= 0 && jj < NCHUNK) ? eo2[((size_t)b * NCHUNK + jj) * HHH + c] : 0.f;
}

__global__ void k_copy_rows(float* __restrict__ dst, int dstStride,
                            const float* __restrict__ src, int srcStride,
                            int rows, int cols) {
  int idx = blockIdx.x * blockDim.x + threadIdx.x;
  if (idx >= rows * cols) return;
  int r = idx / cols, c = idx - r * cols;
  dst[(size_t)r * dstStride + c] = src[(size_t)r * srcStride + c];
}

// Attention + decoder input build. One block (512 thr) per batch element.
// scores = eo2[b]·h ; alpha = softmax ; ctx = alpha·eo2[b]
// xcat[b] = [emb[token] | ctx | h] as f16 (K = 1536)
__global__ __launch_bounds__(512)
void k_dec_attn(const float* __restrict__ eo2, const float* __restrict__ h,
                const float* __restrict__ emb, const int* __restrict__ tv,
                int t, h16* __restrict__ xcat) {
  __shared__ float red[512];
  __shared__ float alpha[NCHUNK];
  int b = blockIdx.x;
  int c = threadIdx.x;
  float hv = h[(size_t)b * HHH + c];

  for (int k = 0; k < NCHUNK; ++k) {
    red[c] = eo2[((size_t)b * NCHUNK + k) * HHH + c] * hv;
    __syncthreads();
    for (int s = 256; s > 0; s >>= 1) {
      if (c < s) red[c] += red[c + s];
      __syncthreads();
    }
    if (c == 0) alpha[k] = red[0];
    __syncthreads();
  }
  if (c == 0) {
    float mx = alpha[0];
    for (int k = 1; k < NCHUNK; ++k) mx = fmaxf(mx, alpha[k]);
    float s = 0.f;
    for (int k = 0; k < NCHUNK; ++k) { alpha[k] = __expf(alpha[k] - mx); s += alpha[k]; }
    float inv = 1.f / s;
    for (int k = 0; k < NCHUNK; ++k) alpha[k] *= inv;
  }
  __syncthreads();

  float ctx = 0.f;
  for (int k = 0; k < NCHUNK; ++k)
    ctx += alpha[k] * eo2[((size_t)b * NCHUNK + k) * HHH + c];

  int tok = tv[b * MAXLENN + t];
  h16* row = xcat + (size_t)b * 1536;
  row[c]        = (h16)emb[(size_t)tok * HHH + c];
  row[512 + c]  = (h16)ctx;
  row[1024 + c] = (h16)hv;
}

// log_softmax over V=16000 and store to d_out[b, t, :]
__global__ __launch_bounds__(256)
void k_logsoftmax(const float* __restrict__ logits, float* __restrict__ out, int t) {
  __shared__ float red[256];
  int b = blockIdx.x, tid = threadIdx.x;
  const float* row = logits + (size_t)b * VVV;
  float mx = -3.4e38f;
  for (int v = tid; v < VVV; v += 256) mx = fmaxf(mx, row[v]);
  red[tid] = mx; __syncthreads();
  for (int s = 128; s > 0; s >>= 1) {
    if (tid < s) red[tid] = fmaxf(red[tid], red[tid + s]);
    __syncthreads();
  }
  float smax = red[0]; __syncthreads();
  float sum = 0.f;
  for (int v = tid; v < VVV; v += 256) sum += __expf(row[v] - smax);
  red[tid] = sum; __syncthreads();
  for (int s = 128; s > 0; s >>= 1) {
    if (tid < s) red[tid] += red[tid + s];
    __syncthreads();
  }
  float lg = smax + __logf(red[0]);
  float* o = out + ((size_t)b * NSTEP + t) * VVV;
  for (int v = tid; v < VVV; v += 256) o[v] = row[v] - lg;
}

// ---------------------------------------------------------------------------
extern "C" void kernel_launch(void* const* d_in, const int* in_sizes, int n_in,
                              void* d_out, int out_size, void* d_ws, size_t ws_size,
                              hipStream_t stream) {
  const float* vid   = (const float*)d_in[0];
  const int*   tv    = (const int*)d_in[1];
  const float* eWih  = (const float*)d_in[2];
  const float* eWhh  = (const float*)d_in[3];
  const float* eb    = (const float*)d_in[4];
  const float* lWih  = (const float*)d_in[5];
  const float* lWhh  = (const float*)d_in[6];
  const float* lb    = (const float*)d_in[7];
  const float* emb   = (const float*)d_in[8];
  const float* dWih  = (const float*)d_in[9];
  const float* dWhh  = (const float*)d_in[10];
  const float* db    = (const float*)d_in[11];
  const float* Wout  = (const float*)d_in[12];
  const float* bout  = (const float*)d_in[13];
  float* out = (float*)d_out;

  // workspace carve-up
  size_t off = 0;
  auto alloc = [&](size_t bytes) {
    void* p = (char*)d_ws + off;
    off += (bytes + 255) & ~(size_t)255;
    return p;
  };
  h16*   vid_h   = (h16*)alloc((size_t)BB * TTT * DVV * 2);    // f16 video feats
  h16*   encW    = (h16*)alloc((size_t)2048 * 1024 * 2);       // [Wih|Whh]
  h16*   l2W     = (h16*)alloc((size_t)2048 * 1024 * 2);
  h16*   decW    = (h16*)alloc((size_t)2048 * 1536 * 2);
  h16*   WoutH   = (h16*)alloc((size_t)VVV * HHH * 2);
  h16*   xcat    = (h16*)alloc((size_t)1280 * 1024 * 2);       // >= 128*1536
  h16*   hEncH   = (h16*)alloc((size_t)1280 * HHH * 2);        // enc h f16 (-> hN)
  h16*   hL2H    = (h16*)alloc((size_t)BB * HHH * 2);
  float* zbuf    = (float*)alloc((size_t)1280 * 2048 * 4);     // z / logits scratch
  float* hEnc    = (float*)alloc((size_t)1280 * HHH * 4);
  float* cEnc    = (float*)alloc((size_t)1280 * HHH * 4);
  float* hL2     = (float*)alloc((size_t)BB * HHH * 4);        // later: decoder h
  float* cL2     = (float*)alloc((size_t)BB * HHH * 4);        // later: decoder c
  float* eo2     = (float*)alloc((size_t)BB * NCHUNK * HHH * 4);
  float* ofirst  = (float*)alloc((size_t)1280 * 4);
  float* fil0    = (float*)alloc(256);
  int*   jbuf    = (int*)alloc((size_t)BB * 4);

  auto cdiv = [](int a, int b) { return (a + b - 1) / b; };
  const int TPB = 256;

  // ---- weight/activation prep (f32 -> f16, concat [Wih|Whh]) ----
  { int n = BB * TTT * DVV;
    k_pack_f16<<<cdiv(n, TPB), TPB, 0, stream>>>(vid_h, vid, 1, n, n, 0); }
  k_pack_f16<<<cdiv(2048 * 512, TPB), TPB, 0, stream>>>(encW, eWih, 2048, 512, 1024, 0);
  k_pack_f16<<<cdiv(2048 * 512, TPB), TPB, 0, stream>>>(encW, eWhh, 2048, 512, 1024, 512);
  k_pack_f16<<<cdiv(2048 * 512, TPB), TPB, 0, stream>>>(l2W, lWih, 2048, 512, 1024, 0);
  k_pack_f16<<<cdiv(2048 * 512, TPB), TPB, 0, stream>>>(l2W, lWhh, 2048, 512, 1024, 512);
  k_pack_f16<<<cdiv(2048 * 1024, TPB), TPB, 0, stream>>>(decW, dWih, 2048, 1024, 1536, 0);
  k_pack_f16<<<cdiv(2048 * 512, TPB), TPB, 0, stream>>>(decW, dWhh, 2048, 512, 1536, 1024);
  { int n = VVV * HHH;
    k_pack_f16<<<cdiv(n, TPB), TPB, 0, stream>>>(WoutH, Wout, 1, n, n, 0); }

  // ---- zero-init states ----
  k_zero_f32<<<cdiv(1280 * HHH, TPB), TPB, 0, stream>>>(hEnc, 1280 * HHH);
  k_zero_f32<<<cdiv(1280 * HHH, TPB), TPB, 0, stream>>>(cEnc, 1280 * HHH);
  k_zero_f16<<<cdiv(1280 * HHH, TPB), TPB, 0, stream>>>(hEncH, 1280 * HHH);
  k_zero_f32<<<cdiv(BB * HHH, TPB), TPB, 0, stream>>>(hL2, BB * HHH);
  k_zero_f32<<<cdiv(BB * HHH, TPB), TPB, 0, stream>>>(cL2, BB * HHH);
  k_zero_f16<<<cdiv(BB * HHH, TPB), TPB, 0, stream>>>(hL2H, BB * HHH);

  // ---- encoder LSTM: 1280 sequences x 8 steps ----
  for (int t = 0; t < CLL; ++t) {
    // xcat[1280,1024] = [vid_h[:, t, :] | h]
    k_concat2<<<cdiv(1280 * 1024, TPB), TPB, 0, stream>>>(
        xcat, vid_h + (size_t)t * DVV, CLL * DVV, hEncH, HHH, 1280, DVV, HHH);
    // z = xcat @ encW.T + eb   (M=1280, N=2048, K=1024)
    k_gemm_wmma<<<dim3(2048 / 128, 1280 / 64), 256, 0, stream>>>(
        xcat, encW, eb, zbuf, 1280, 2048, 1024);
    k_lstm_cell<<<cdiv(1280 * HHH, TPB), TPB, 0, stream>>>(zbuf, cEnc, hEnc, hEncH, 1280);
    if (t == 0)
      k_snap_ofirst<<<cdiv(1280, TPB), TPB, 0, stream>>>(ofirst, hEnc, 1280);
  }
  // hEncH now holds hN (f16) = l2 input, laid out [B, CHUNKS, H]

  // ---- layer-2 LSTM: 128 x 10 steps ----
  for (int t = 0; t < NCHUNK; ++t) {
    k_concat2<<<cdiv(BB * 1024, TPB), TPB, 0, stream>>>(
        xcat, hEncH + (size_t)t * HHH, NCHUNK * HHH, hL2H, HHH, BB, HHH, HHH);
    k_gemm_wmma<<<dim3(2048 / 128, BB / 64), 256, 0, stream>>>(
        xcat, l2W, lb, zbuf, BB, 2048, 1024);
    k_lstm_cell<<<cdiv(BB * HHH, TPB), TPB, 0, stream>>>(zbuf, cL2, hL2, hL2H, BB);
    k_copy_rows<<<cdiv(BB * HHH, TPB), TPB, 0, stream>>>(
        eo2 + (size_t)t * HHH, NCHUNK * HHH, hL2, HHH, BB, HHH);
  }
  // cL2 == c_dec already

  // ---- chunk-mask selection, h_dec gather ----
  k_fil0<<<1, 32, 0, stream>>>(eb, fil0);
  k_select_j<<<cdiv(BB, TPB), TPB, 0, stream>>>(ofirst, fil0, jbuf);
  k_gather_hdec<<<cdiv(BB * HHH, TPB), TPB, 0, stream>>>(eo2, jbuf, hL2);

  // ---- attention decoder: 27 steps ----
  for (int t = 0; t < NSTEP; ++t) {
    // attention + build xcat[b] = [emb[token] | ctx | h] (f16, K=1536)
    k_dec_attn<<<BB, 512, 0, stream>>>(eo2, hL2, emb, tv, t, xcat);
    // z = xcat @ decW.T + db   (M=128, N=2048, K=1536)
    k_gemm_wmma<<<dim3(2048 / 128, BB / 64), 256, 0, stream>>>(
        xcat, decW, db, zbuf, BB, 2048, 1536);
    k_lstm_cell<<<cdiv(BB * HHH, TPB), TPB, 0, stream>>>(zbuf, cL2, hL2, hL2H, BB);
    // logits = h @ Wout.T + bout   (M=128, N=16000, K=512)
    k_gemm_wmma<<<dim3(VVV / 128, BB / 64), 256, 0, stream>>>(
        hL2H, WoutH, bout, zbuf, BB, VVV, HHH);
    k_logsoftmax<<<BB, 256, 0, stream>>>(zbuf, out, t);
  }
}
